// MPNEncoder_39960375722520
// MI455X (gfx1250) — matrix operand order repaired and checked by reading, not compiled, not run on hardware
//
#include <hip/hip_runtime.h>

// ---------------------------------------------------------------------------
// MPNEncoder (GRU message passing) for MI455X / gfx1250, wave32 + WMMA bf16.
//
// Restructuring vs reference:
//   * zx = fmess@Wz[:384], hx = fmess@Wh[:384], rx = fmess@Wr, ox = fnode@Wo[:256]
//     precomputed once (loop-invariant).
//   * h_nei @ Ur == (h @ Ur)[bgraph]  -> one GEMM per step instead of four.
// GEMMs run as bf16 x bf16 -> f32 via v_wmma_f32_16x16x32_bf16; all state
// (h, sum_h, gate pre-activations) stays fp32.
// ---------------------------------------------------------------------------

#define NN   40000    // nodes
#define NE   80000    // edges
#define NB   4        // max neighbors
#define ISZ  384      // INPUT_SIZE
#define HID  256      // HIDDEN == NODE_FDIM
#define DEPTH 6       // reference constant (device scalar can't steer capture)

typedef __attribute__((ext_vector_type(16))) __bf16        bf16x16;
typedef __attribute__((ext_vector_type(8)))  float         f32x8;
typedef __attribute__((ext_vector_type(4)))  float         f32x4;
typedef __attribute__((ext_vector_type(4)))  unsigned int  u32x4;
typedef __attribute__((ext_vector_type(4)))  unsigned short u16x4;

union FragBF { bf16x16 v; u32x4 q[2]; };

__device__ __forceinline__ unsigned short f2bf(float x) {
    unsigned int u = __float_as_uint(x);
    unsigned int r = u + 0x7FFFu + ((u >> 16) & 1u);   // round-to-nearest-even
    return (unsigned short)(r >> 16);
}
__device__ __forceinline__ float sigmoidf_(float x) { return 1.0f / (1.0f + __expf(-x)); }

// ------------------------------- small utils -------------------------------
__global__ void cvt_bf16_kernel(const float* __restrict__ src,
                                unsigned short* __restrict__ dst, long long n) {
    long long t = (long long)blockIdx.x * blockDim.x + threadIdx.x;
    if (t < n) dst[t] = f2bf(src[t]);
}

// Bt[n][k] = bf16(W[rowStart + k][n])  (pack B transposed, bf16)
__global__ void pack_w_kernel(const float* __restrict__ W, int rowStart, int K,
                              unsigned short* __restrict__ Bt) {
    long long t = (long long)blockIdx.x * blockDim.x + threadIdx.x;
    if (t >= (long long)HID * K) return;
    int n = (int)(t / K);
    int k = (int)(t % K);
    Bt[(size_t)n * K + k] = f2bf(W[(size_t)(rowStart + k) * HID + n]);
}

__global__ void zero_u32_kernel(unsigned int* __restrict__ p, long long n) {
    long long t = (long long)blockIdx.x * blockDim.x + threadIdx.x;
    if (t < n) p[t] = 0u;
}

// ----------------------------- WMMA GEMM (N=256) ----------------------------
// C[M][256] = A[M][K] (bf16, row major) @ B[K][256] given as Bt[256][K] bf16.
// Block: 256 thr (8 waves). BM=128, waves 2(M) x 4(N), each wave 64x64 tile.
#define BM   128
#define KC   128                 // K elements staged in LDS per chunk
#define LPAD 8
#define LSTR (KC + LPAD)         // 136 ushorts per LDS row

__global__ __launch_bounds__(256) void gemm_bf16_n256_kernel(
    const unsigned short* __restrict__ A,
    const unsigned short* __restrict__ Bt,
    float* __restrict__ C, int M, int K)
{
    extern __shared__ unsigned short ldsB[];   // 256 * LSTR ushorts

    const int tid  = threadIdx.x;
    const int lane = tid & 31;
    const int wave = tid >> 5;
    const int wm   = wave & 1;
    const int wn   = wave >> 1;
    const int half = lane >> 4;
    const int l16  = lane & 15;

    const int rowBase = blockIdx.x * BM + wm * 64;
    const int colBase = wn * 64;

    f32x8 zero;
    #pragma unroll
    for (int v = 0; v < 8; ++v) zero[v] = 0.0f;

    f32x8 c[4][4];
    #pragma unroll
    for (int i = 0; i < 4; ++i)
        #pragma unroll
        for (int j = 0; j < 4; ++j) c[i][j] = zero;

    for (int kc = 0; kc < K; kc += KC) {
        __syncthreads();
        // stage Bt[:, kc:kc+KC] into LDS (256 rows x 128 bf16, padded rows)
        for (int idx = tid; idx < 256 * (KC / 8); idx += 256) {
            int r  = idx >> 4;        // KC/8 == 16 chunks per row
            int kq = idx & 15;
            *(u32x4*)&ldsB[r * LSTR + kq * 8] =
                *(const u32x4*)&Bt[(size_t)r * K + kc + kq * 8];
        }
        __syncthreads();

        #pragma unroll
        for (int ks = 0; ks < KC; ks += 32) {
            const int k0 = kc + ks;
            // A fragments: 16-bit A layout -> lane half owns {k, k+16} chunks
            FragBF a[4];
            #pragma unroll
            for (int i = 0; i < 4; ++i) {
                int row = rowBase + i * 16 + l16;
                if (row >= M) row = M - 1;
                const unsigned short* ap = A + (size_t)row * K + k0 + half * 8;
                a[i].q[0] = *(const u32x4*)(ap);
                a[i].q[1] = *(const u32x4*)(ap + 16);
            }
            // B fragments: lane half owns {k, k+16} chunks (same as A)
            FragBF b[4];
            #pragma unroll
            for (int j = 0; j < 4; ++j) {
                const unsigned short* bp =
                    ldsB + (colBase + j * 16 + l16) * LSTR + ks + half * 8;
                b[j].q[0] = *(const u32x4*)(bp);
                b[j].q[1] = *(const u32x4*)(bp + 16);
            }
            #pragma unroll
            for (int i = 0; i < 4; ++i)
                #pragma unroll
                for (int j = 0; j < 4; ++j)
                    c[i][j] = __builtin_amdgcn_wmma_f32_16x16x32_bf16(
                        false, a[i].v, false, b[j].v, (short)0, c[i][j],
                        false, false);
        }
    }

    // C layout: lanes 0-15 -> rows 0-7 (VGPR v), lanes 16-31 -> rows 8-15
    #pragma unroll
    for (int i = 0; i < 4; ++i) {
        int rbase = rowBase + i * 16 + half * 8;
        #pragma unroll
        for (int j = 0; j < 4; ++j) {
            int col = colBase + j * 16 + l16;
            #pragma unroll
            for (int v = 0; v < 8; ++v) {
                int row = rbase + v;
                if (row < M) C[(size_t)row * 256 + col] = c[i][j][v];
            }
        }
    }
}

// ------------------------ gather / elementwise kernels ----------------------
// sum_h[e] = sum_k h[bgraph[e,k]] ; also bf16 copy for the Wz GEMM
__global__ void sumh_kernel(const float* __restrict__ h,
                            const int* __restrict__ bgraph,
                            float* __restrict__ sum_h,
                            unsigned short* __restrict__ sumh_bf) {
    long long t = (long long)blockIdx.x * blockDim.x + threadIdx.x;
    if (t >= (long long)NE * 64) return;
    int e  = (int)(t >> 6);
    int c4 = (int)(t & 63) * 4;
    f32x4 acc; acc[0] = acc[1] = acc[2] = acc[3] = 0.0f;
    #pragma unroll
    for (int k = 0; k < NB; ++k) {
        int nb = bgraph[e * NB + k];
        f32x4 v = *(const f32x4*)&h[(size_t)nb * HID + c4];
        acc += v;
    }
    *(f32x4*)&sum_h[(size_t)e * HID + c4] = acc;
    u16x4 b; b[0] = f2bf(acc[0]); b[1] = f2bf(acc[1]);
    b[2] = f2bf(acc[2]); b[3] = f2bf(acc[3]);
    *(u16x4*)&sumh_bf[(size_t)e * HID + c4] = b;
}

// sum_gh[e] = sum_k sigmoid(rx[e] + hu[bgraph[e,k]] + bur) * h[bgraph[e,k]]
__global__ void sumgh_kernel(const float* __restrict__ h,
                             const float* __restrict__ hu,
                             const float* __restrict__ rx,
                             const float* __restrict__ bur,
                             const int* __restrict__ bgraph,
                             unsigned short* __restrict__ sumgh_bf) {
    long long t = (long long)blockIdx.x * blockDim.x + threadIdx.x;
    if (t >= (long long)NE * 64) return;
    int e  = (int)(t >> 6);
    int c4 = (int)(t & 63) * 4;
    f32x4 rxv  = *(const f32x4*)&rx[(size_t)e * HID + c4];
    f32x4 burv = *(const f32x4*)&bur[c4];
    f32x4 acc; acc[0] = acc[1] = acc[2] = acc[3] = 0.0f;
    #pragma unroll
    for (int k = 0; k < NB; ++k) {
        int nb = bgraph[e * NB + k];
        f32x4 hv  = *(const f32x4*)&h[(size_t)nb * HID + c4];
        f32x4 huv = *(const f32x4*)&hu[(size_t)nb * HID + c4];
        #pragma unroll
        for (int v = 0; v < 4; ++v)
            acc[v] += sigmoidf_(rxv[v] + huv[v] + burv[v]) * hv[v];
    }
    u16x4 b; b[0] = f2bf(acc[0]); b[1] = f2bf(acc[1]);
    b[2] = f2bf(acc[2]); b[3] = f2bf(acc[3]);
    *(u16x4*)&sumgh_bf[(size_t)e * HID + c4] = b;
}

// h_new = ((1-z)*sum_h + z*pre_h) * msg_mask ; also write bf16 copy
__global__ void combine_kernel(const float* __restrict__ zx,
                               const float* __restrict__ zpre,
                               const float* __restrict__ bz,
                               const float* __restrict__ hx,
                               const float* __restrict__ hpre,
                               const float* __restrict__ bh,
                               const float* __restrict__ sum_h,
                               float* __restrict__ h,
                               unsigned short* __restrict__ h_bf) {
    long long t = (long long)blockIdx.x * blockDim.x + threadIdx.x;
    if (t >= (long long)NE * 64) return;
    int e  = (int)(t >> 6);
    int c4 = (int)(t & 63) * 4;
    size_t o = (size_t)e * HID + c4;
    f32x4 zxv = *(const f32x4*)&zx[o];
    f32x4 zpv = *(const f32x4*)&zpre[o];
    f32x4 bzv = *(const f32x4*)&bz[c4];
    f32x4 hxv = *(const f32x4*)&hx[o];
    f32x4 hpv = *(const f32x4*)&hpre[o];
    f32x4 bhv = *(const f32x4*)&bh[c4];
    f32x4 shv = *(const f32x4*)&sum_h[o];
    f32x4 hn;
    #pragma unroll
    for (int v = 0; v < 4; ++v) {
        float z  = sigmoidf_(zxv[v] + zpv[v] + bzv[v]);
        float ph = tanhf(hxv[v] + hpv[v] + bhv[v]);
        hn[v] = (1.0f - z) * shv[v] + z * ph;
    }
    if (e == 0) { hn[0] = hn[1] = hn[2] = hn[3] = 0.0f; }  // msg_mask row 0
    *(f32x4*)&h[o] = hn;
    u16x4 b; b[0] = f2bf(hn[0]); b[1] = f2bf(hn[1]);
    b[2] = f2bf(hn[2]); b[3] = f2bf(hn[3]);
    *(u16x4*)&h_bf[o] = b;
}

// nei_bf[n] = bf16( sum_k h[agraph[n,k]] )
__global__ void neisum_kernel(const float* __restrict__ h,
                              const int* __restrict__ agraph,
                              unsigned short* __restrict__ nei_bf) {
    long long t = (long long)blockIdx.x * blockDim.x + threadIdx.x;
    if (t >= (long long)NN * 64) return;
    int n  = (int)(t >> 6);
    int c4 = (int)(t & 63) * 4;
    f32x4 acc; acc[0] = acc[1] = acc[2] = acc[3] = 0.0f;
    #pragma unroll
    for (int k = 0; k < NB; ++k) {
        int nb = agraph[n * NB + k];
        acc += *(const f32x4*)&h[(size_t)nb * HID + c4];
    }
    u16x4 b; b[0] = f2bf(acc[0]); b[1] = f2bf(acc[1]);
    b[2] = f2bf(acc[2]); b[3] = f2bf(acc[3]);
    *(u16x4*)&nei_bf[(size_t)n * HID + c4] = b;
}

// out = relu(ox + op + bo) * mask
__global__ void out_kernel(const float* __restrict__ ox,
                           const float* __restrict__ op,
                           const float* __restrict__ bo,
                           const float* __restrict__ mask,
                           float* __restrict__ out) {
    long long t = (long long)blockIdx.x * blockDim.x + threadIdx.x;
    if (t >= (long long)NN * 64) return;
    int n  = (int)(t >> 6);
    int c4 = (int)(t & 63) * 4;
    size_t o = (size_t)n * HID + c4;
    f32x4 a = *(const f32x4*)&ox[o];
    f32x4 b = *(const f32x4*)&op[o];
    f32x4 c = *(const f32x4*)&bo[c4];
    float m = mask[n];
    f32x4 r;
    #pragma unroll
    for (int v = 0; v < 4; ++v) {
        float x = a[v] + b[v] + c[v];
        r[v] = (x > 0.0f ? x : 0.0f) * m;
    }
    *(f32x4*)&out[o] = r;
}

// --------------------------------- launch -----------------------------------
extern "C" void kernel_launch(void* const* d_in, const int* in_sizes, int n_in,
                              void* d_out, int out_size, void* d_ws, size_t ws_size,
                              hipStream_t stream) {
    (void)in_sizes; (void)n_in; (void)out_size; (void)ws_size;

    const float* fnode  = (const float*)d_in[0];
    const float* fmess  = (const float*)d_in[1];
    const int*   agraph = (const int*)d_in[2];
    const int*   bgraph = (const int*)d_in[3];
    const float* mask   = (const float*)d_in[4];
    const float* Wz     = (const float*)d_in[5];
    const float* bz     = (const float*)d_in[6];
    const float* Wr     = (const float*)d_in[7];
    const float* Ur     = (const float*)d_in[8];
    const float* bur    = (const float*)d_in[9];
    const float* Wh     = (const float*)d_in[10];
    const float* bh     = (const float*)d_in[11];
    const float* Wo     = (const float*)d_in[12];
    const float* bo     = (const float*)d_in[13];
    // d_in[14] = depth (==DEPTH, fixed by the reference)

    // ---- workspace carve-up (256B aligned) ----
    unsigned char* base = (unsigned char*)d_ws;
    size_t off = 0;
    auto carve = [&](size_t bytes) -> void* {
        void* p = base + off;
        off = (off + bytes + 255) & ~(size_t)255;
        return p;
    };
    const size_t EH4 = (size_t)NE * HID * 4, EH2 = (size_t)NE * HID * 2;
    const size_t NH4 = (size_t)NN * HID * 4, NH2 = (size_t)NN * HID * 2;

    unsigned short* fmess_bf = (unsigned short*)carve((size_t)NE * ISZ * 2);
    unsigned short* fnode_bf = (unsigned short*)carve(NH2);
    unsigned short* WrT    = (unsigned short*)carve((size_t)HID * ISZ * 2);
    unsigned short* WzTopT = (unsigned short*)carve((size_t)HID * ISZ * 2);
    unsigned short* WzBotT = (unsigned short*)carve((size_t)HID * HID * 2);
    unsigned short* WhTopT = (unsigned short*)carve((size_t)HID * ISZ * 2);
    unsigned short* WhBotT = (unsigned short*)carve((size_t)HID * HID * 2);
    unsigned short* UrT    = (unsigned short*)carve((size_t)HID * HID * 2);
    unsigned short* WoTopT = (unsigned short*)carve((size_t)HID * HID * 2);
    unsigned short* WoBotT = (unsigned short*)carve((size_t)HID * HID * 2);
    float* rx    = (float*)carve(EH4);
    float* zx    = (float*)carve(EH4);
    float* hx    = (float*)carve(EH4);
    float* ox    = (float*)carve(NH4);
    float* h     = (float*)carve(EH4);
    unsigned short* h_bf = (unsigned short*)carve(EH2);
    float* sum_h = (float*)carve(EH4);
    unsigned short* sumh_bf  = (unsigned short*)carve(EH2);
    float* bufA  = (float*)carve(EH4);   // hu, then zpre (live ranges disjoint)
    float* bufB  = (float*)carve(EH4);   // hpre
    unsigned short* sumgh_bf = (unsigned short*)carve(EH2);
    unsigned short* nei_bf   = (unsigned short*)carve(NH2);
    float* op    = (float*)carve(NH4);

    auto cdiv = [](long long a, long long b) { return (unsigned)((a + b - 1) / b); };
    const unsigned T = 256;
    const size_t ldsBytes = 256 * LSTR * sizeof(unsigned short);
    const unsigned gE  = cdiv(NE, BM);              // 625
    const unsigned gN  = cdiv(NN, BM);              // 313
    const unsigned eEW = cdiv((long long)NE * 64, T);
    const unsigned eNW = cdiv((long long)NN * 64, T);

    // ---- precompute ----
    cvt_bf16_kernel<<<cdiv((long long)NE * ISZ, T), T, 0, stream>>>(fmess, fmess_bf, (long long)NE * ISZ);
    cvt_bf16_kernel<<<cdiv((long long)NN * HID, T), T, 0, stream>>>(fnode, fnode_bf, (long long)NN * HID);

    pack_w_kernel<<<cdiv((long long)HID * ISZ, T), T, 0, stream>>>(Wr, 0,   ISZ, WrT);
    pack_w_kernel<<<cdiv((long long)HID * ISZ, T), T, 0, stream>>>(Wz, 0,   ISZ, WzTopT);
    pack_w_kernel<<<cdiv((long long)HID * HID, T), T, 0, stream>>>(Wz, ISZ, HID, WzBotT);
    pack_w_kernel<<<cdiv((long long)HID * ISZ, T), T, 0, stream>>>(Wh, 0,   ISZ, WhTopT);
    pack_w_kernel<<<cdiv((long long)HID * HID, T), T, 0, stream>>>(Wh, ISZ, HID, WhBotT);
    pack_w_kernel<<<cdiv((long long)HID * HID, T), T, 0, stream>>>(Ur, 0,   HID, UrT);
    pack_w_kernel<<<cdiv((long long)HID * HID, T), T, 0, stream>>>(Wo, 0,   HID, WoTopT);
    pack_w_kernel<<<cdiv((long long)HID * HID, T), T, 0, stream>>>(Wo, HID, HID, WoBotT);

    gemm_bf16_n256_kernel<<<gE, T, ldsBytes, stream>>>(fmess_bf, WrT,    rx, NE, ISZ);
    gemm_bf16_n256_kernel<<<gE, T, ldsBytes, stream>>>(fmess_bf, WzTopT, zx, NE, ISZ);
    gemm_bf16_n256_kernel<<<gE, T, ldsBytes, stream>>>(fmess_bf, WhTopT, hx, NE, ISZ);
    gemm_bf16_n256_kernel<<<gN, T, ldsBytes, stream>>>(fnode_bf, WoTopT, ox, NN, HID);

    zero_u32_kernel<<<cdiv((long long)NE * HID, T), T, 0, stream>>>((unsigned int*)h, (long long)NE * HID);
    zero_u32_kernel<<<cdiv((long long)NE * HID / 2, T), T, 0, stream>>>((unsigned int*)h_bf, (long long)NE * HID / 2);

    // ---- GRU steps ----
    for (int d = 0; d < DEPTH; ++d) {
        sumh_kernel<<<eEW, T, 0, stream>>>(h, bgraph, sum_h, sumh_bf);
        gemm_bf16_n256_kernel<<<gE, T, ldsBytes, stream>>>(h_bf, UrT, bufA, NE, HID);      // hu
        sumgh_kernel<<<eEW, T, 0, stream>>>(h, bufA, rx, bur, bgraph, sumgh_bf);
        gemm_bf16_n256_kernel<<<gE, T, ldsBytes, stream>>>(sumh_bf,  WzBotT, bufA, NE, HID); // zpre
        gemm_bf16_n256_kernel<<<gE, T, ldsBytes, stream>>>(sumgh_bf, WhBotT, bufB, NE, HID); // hpre
        combine_kernel<<<eEW, T, 0, stream>>>(zx, bufA, bz, hx, bufB, bh, sum_h, h, h_bf);
    }

    // ---- readout ----
    neisum_kernel<<<eNW, T, 0, stream>>>(h, agraph, nei_bf);
    gemm_bf16_n256_kernel<<<gN, T, ldsBytes, stream>>>(nei_bf, WoBotT, op, NN, HID);
    out_kernel<<<eNW, T, 0, stream>>>(ox, op, bo, mask, (float*)d_out);

    // second tuple output: final h
    hipMemcpyAsync((float*)d_out + (size_t)NN * HID, h, EH4,
                   hipMemcpyDeviceToDevice, stream);
}
